// SimpleTransformerBlock_28140625723543
// MI455X (gfx1250) — matrix operand order repaired and checked by reading, not compiled, not run on hardware
//
#include <hip/hip_runtime.h>

// ---------------- types / helpers ----------------
typedef unsigned short u16;
typedef __attribute__((ext_vector_type(16))) __bf16 bf16x16;
typedef __attribute__((ext_vector_type(8)))  float  f32x8;
typedef __attribute__((ext_vector_type(4)))  unsigned int u32x4;
typedef __attribute__((ext_vector_type(8)))  int i32x8;
typedef __attribute__((ext_vector_type(4)))  int i32x4;

union ABFrag { bf16x16 v; uint4 q[2]; };
union CFrag  { f32x8  v; float f[8]; };

__device__ __forceinline__ u16 f2bf(float f) {
  unsigned u = __float_as_uint(f);
  u += 0x7FFFu + ((u >> 16) & 1u);          // round-to-nearest-even
  return (u16)(u >> 16);
}
__device__ __forceinline__ float bf2f(u16 h) {
  return __uint_as_float(((unsigned)h) << 16);
}

#define DMODEL 2048
#define NROWS  4096      // B*S = 2*2048
#define DINTER 8192
#define NHEADS 16
#define HEADD  128
#define ATTN_SCALE 0.08838834764831845f  // 1/sqrt(128)

// ---------------- TDM (Tensor Data Mover) async tile load ----------------
#if __has_builtin(__builtin_amdgcn_tensor_load_to_lds) && __has_builtin(__builtin_amdgcn_s_wait_tensorcnt)
#define HAVE_TDM 1
#pragma message("CDNA5: HAVE_TDM=1 (tensor_load_to_lds + s_wait_tensorcnt available)")
#else
#define HAVE_TDM 0
#pragma message("CDNA5: HAVE_TDM=0 (falling back to register-prefetch staging)")
#endif

#if HAVE_TDM
// 2D bf16 tile load: tile (tx elems contiguous) x (ty rows), row stride strideElems,
// into contiguous LDS at `lds`. Descriptor per CDNA5 ISA section 8.3/8.4.
__device__ __forceinline__ void tdm_load_2d(const void* lds, const u16* g,
                                            unsigned tx, unsigned ty,
                                            unsigned long long strideElems) {
  unsigned ldsOff = (unsigned)(unsigned long long)lds;      // generic addr low 32 = LDS offset
  unsigned long long ga = (unsigned long long)g;
  u32x4 g0;
  g0[0] = 1u;                                               // count=1 (valid), user mode
  g0[1] = ldsOff;                                           // lds_addr
  g0[2] = (unsigned)(ga & 0xFFFFFFFFu);                     // global_addr[31:0]
  g0[3] = (unsigned)((ga >> 32) & 0x01FFFFFFu) | (2u << 30); // global_addr[56:32], type=2
  i32x8 g1;
  g1[0] = (int)(1u << 16);                                  // data_size=1 -> 2 bytes
  g1[1] = (int)((tx & 0xFFFFu) << 16);                      // tensor_dim0[15:0]
  g1[2] = (int)(((tx >> 16) & 0xFFFFu) | ((ty & 0xFFFFu) << 16));   // dim0 hi | dim1 lo
  g1[3] = (int)(((ty >> 16) & 0xFFFFu) | ((tx & 0xFFFFu) << 16));   // dim1 hi | tile_dim0
  g1[4] = (int)(ty & 0xFFFFu);                              // tile_dim1 (tile_dim2 = 0)
  g1[5] = (int)(strideElems & 0xFFFFFFFFu);                 // tensor_dim0_stride lo
  g1[6] = (int)((strideElems >> 32) & 0xFFFFu);             // stride hi (dim1_stride = 0)
  g1[7] = 0;
  i32x4 z4 = {0, 0, 0, 0};
#if defined(__clang_major__) && (__clang_major__ >= 23)
  i32x8 z8 = {0, 0, 0, 0, 0, 0, 0, 0};
  __builtin_amdgcn_tensor_load_to_lds(g0, g1, z4, z4, z8, 0);
#else
  __builtin_amdgcn_tensor_load_to_lds(g0, g1, z4, z4, 0);
#endif
}
#endif

// ---------------- fp32 [K][N] -> bf16 transposed [N][K] (LDS tiled) ----------------
__global__ __launch_bounds__(256) void cvt_t_kernel(const float* __restrict__ src,
                                                    u16* __restrict__ dst,
                                                    int Kd, int Nd) {
  __shared__ u16 tile[32][33];            // +1 pad: avoid bank conflicts
  const int tid = threadIdx.x;
  const int k0 = blockIdx.y * 32;
  const int n0 = blockIdx.x * 32;
  {
    int r = tid >> 3, c = (tid & 7) * 4;
    float4 f = *(const float4*)(src + (size_t)(k0 + r) * Nd + n0 + c);
    tile[r][c + 0] = f2bf(f.x);
    tile[r][c + 1] = f2bf(f.y);
    tile[r][c + 2] = f2bf(f.z);
    tile[r][c + 3] = f2bf(f.w);
  }
  __syncthreads();
  {
    int n = tid >> 3, c = (tid & 7) * 4;
    union { uint2 v; u16 h[4]; } o;
    o.h[0] = tile[c + 0][n];
    o.h[1] = tile[c + 1][n];
    o.h[2] = tile[c + 2][n];
    o.h[3] = tile[c + 3][n];
    *(uint2*)(dst + (size_t)(n0 + n) * Kd + k0 + c) = o.v;
  }
}

// ---------------- LayerNorm (f32 in, bf16 out), one block per row ----------------
__global__ __launch_bounds__(256) void ln_kernel(const float* __restrict__ x,
                                                 const float* __restrict__ g,
                                                 const float* __restrict__ b,
                                                 u16* __restrict__ out) {
  __shared__ float2 red[256];
  const int tid = threadIdx.x;
  const size_t row = blockIdx.x;
  const float* xr = x + row * DMODEL;
  float v[8];
  float s = 0.f, ss = 0.f;
#pragma unroll
  for (int i = 0; i < 8; i++) {
    v[i] = xr[tid + i * 256];
    s += v[i]; ss += v[i] * v[i];
  }
  red[tid] = make_float2(s, ss);
  __syncthreads();
  for (int st = 128; st > 0; st >>= 1) {
    if (tid < st) {
      red[tid].x += red[tid + st].x;
      red[tid].y += red[tid + st].y;
    }
    __syncthreads();
  }
  const float mean = red[0].x * (1.f / DMODEL);
  const float var  = red[0].y * (1.f / DMODEL) - mean * mean;
  const float rstd = rsqrtf(var + 1e-6f);
  u16* orow = out + row * DMODEL;
#pragma unroll
  for (int i = 0; i < 8; i++) {
    int c = tid + i * 256;
    orow[c] = f2bf((v[i] - mean) * rstd * g[c] + b[c]);
  }
}

// ---------------- tiled bf16 WMMA GEMM, 128x256 block tile, 64x64 wave tile ----------------
// B is PRE-TRANSPOSED: Bt is [N][K] row-major, so both tiles stage contiguously.
// Double-buffered LDS; TDM (tensor_load_to_lds) staging when available, register-
// prefetch pipeline otherwise. 16 WMMAs per 8 fragment loads per wave per K-step.
// MODE 0: outb = bf16(acc)
// MODE 1: outf = resid + acc            (fp32 residual add)
// MODE 2: outb = bf16(silu(acc))
// MODE 3: outb = bf16(acc * bf2f(aux))  (gate*up fuse)
template <int MODE>
__global__ __launch_bounds__(256) void gemm_kernel(const u16* __restrict__ A,
                                                   const u16* __restrict__ Bt,
                                                   int M, int N, int K,
                                                   const float* __restrict__ resid,
                                                   const u16* __restrict__ aux,
                                                   float* __restrict__ outf,
                                                   u16* __restrict__ outb) {
  (void)M;
  __shared__ __align__(16) u16 sA[2][128 * 32];   // [m][k]  16 KB
  __shared__ __align__(16) u16 sB[2][256 * 32];   // [n][k]  32 KB
  const int tid = threadIdx.x;
  const int wave = tid >> 5, lane = tid & 31;
  const int laneHalf = lane & 15, hi = lane >> 4;
  const int wm = wave >> 2, wn = wave & 3;           // waves: 2 x 4 -> 64x64 each
  const size_t bm = (size_t)blockIdx.y * 128;
  const size_t bn = (size_t)blockIdx.x * 256;
  const int nk = K >> 5;

  CFrag acc[4][4];
#pragma unroll
  for (int mf = 0; mf < 4; mf++)
#pragma unroll
    for (int nf = 0; nf < 4; nf++)
#pragma unroll
      for (int j = 0; j < 8; j++) acc[mf][nf].f[j] = 0.f;

  auto compute = [&](int cur) {
    ABFrag af[4], bfr[4];
#pragma unroll
    for (int mf = 0; mf < 4; mf++) {   // A-layout: lo lanes K{0..7,16..23}, hi lanes K{8..15,24..31}
      const u16* p = &sA[cur][(wm * 64 + mf * 16 + laneHalf) * 32 + hi * 8];
      af[mf].q[0] = *(const uint4*)p;
      af[mf].q[1] = *(const uint4*)(p + 16);
    }
#pragma unroll
    for (int nf = 0; nf < 4; nf++) {   // B-layout: lo lanes K{0..7,16..23}, hi lanes K{8..15,24..31}
      const u16* p = &sB[cur][(wn * 64 + nf * 16 + laneHalf) * 32 + hi * 8];
      bfr[nf].q[0] = *(const uint4*)p;
      bfr[nf].q[1] = *(const uint4*)(p + 16);
    }
#pragma unroll
    for (int mf = 0; mf < 4; mf++)
#pragma unroll
      for (int nf = 0; nf < 4; nf++)
        acc[mf][nf].v = __builtin_amdgcn_wmma_f32_16x16x32_bf16(
            false, af[mf].v, false, bfr[nf].v, (short)0, acc[mf][nf].v, false, false);
  };

#if HAVE_TDM
  if (tid < 32) {                      // wave 0 drives the DMA for the block
    tdm_load_2d(&sA[0][0], A + bm * (size_t)K, 32, 128, (unsigned long long)K);
    tdm_load_2d(&sB[0][0], Bt + bn * (size_t)K, 32, 256, (unsigned long long)K);
    __builtin_amdgcn_s_wait_tensorcnt(0);
  }
  __syncthreads();
  for (int kb = 0; kb < nk; kb++) {
    const int cur = kb & 1;
    if ((kb + 1 < nk) && (tid < 32)) {             // async-prefetch next tiles
      const size_t k0 = (size_t)(kb + 1) << 5;
      tdm_load_2d(&sA[cur ^ 1][0], A + bm * (size_t)K + k0, 32, 128, (unsigned long long)K);
      tdm_load_2d(&sB[cur ^ 1][0], Bt + bn * (size_t)K + k0, 32, 256, (unsigned long long)K);
    }
    compute(cur);
    if (tid < 32) __builtin_amdgcn_s_wait_tensorcnt(0);
    __syncthreads();
  }
#else
  {
    const int r = tid >> 1, c = (tid & 1) * 16;
    const u16* asrc = A + (bm + r) * (size_t)K + c;
    *(uint4*)&sA[0][r * 32 + c]     = *(const uint4*)asrc;
    *(uint4*)&sA[0][r * 32 + c + 8] = *(const uint4*)(asrc + 8);
    const u16* bsrc = Bt + (bn + tid) * (size_t)K;
#pragma unroll
    for (int q = 0; q < 4; q++)
      *(uint4*)&sB[0][tid * 32 + q * 8] = *(const uint4*)(bsrc + q * 8);
  }
  __syncthreads();
  for (int kb = 0; kb < nk; kb++) {
    const int cur = kb & 1;
    const bool hasNext = (kb + 1 < nk);
    const int r = tid >> 1, c = (tid & 1) * 16;
    uint4 pa0 = {}, pa1 = {}, pb[4] = {};
    if (hasNext) {                                 // prefetch to registers
      const size_t k0 = (size_t)(kb + 1) << 5;
      const u16* asrc = A + (bm + r) * (size_t)K + k0 + c;
      pa0 = *(const uint4*)asrc; pa1 = *(const uint4*)(asrc + 8);
      const u16* bsrc = Bt + (bn + tid) * (size_t)K + k0;
#pragma unroll
      for (int q = 0; q < 4; q++) pb[q] = *(const uint4*)(bsrc + q * 8);
    }
    compute(cur);
    if (hasNext) {
      const int nxt = cur ^ 1;
      *(uint4*)&sA[nxt][r * 32 + c]     = pa0;
      *(uint4*)&sA[nxt][r * 32 + c + 8] = pa1;
#pragma unroll
      for (int q = 0; q < 4; q++) *(uint4*)&sB[nxt][tid * 32 + q * 8] = pb[q];
    }
    __syncthreads();
  }
#endif

  // epilogue: C layout -> row = j + 8*hi, col = laneHalf within 16x16 tile
#pragma unroll
  for (int mf = 0; mf < 4; mf++)
#pragma unroll
    for (int nf = 0; nf < 4; nf++)
#pragma unroll
      for (int j = 0; j < 8; j++) {
        size_t row = bm + wm * 64 + mf * 16 + j + hi * 8;
        size_t col = bn + wn * 64 + nf * 16 + laneHalf;
        size_t idx = row * (size_t)N + col;
        float v = acc[mf][nf].f[j];
        if (MODE == 0) {
          outb[idx] = f2bf(v);
        } else if (MODE == 1) {
          outf[idx] = resid[idx] + v;
        } else if (MODE == 2) {
          float sg = 1.f / (1.f + __expf(-v));
          outb[idx] = f2bf(v * sg);
        } else {
          outb[idx] = f2bf(v * bf2f(aux[idx]));
        }
      }
}

// ---------------- flash attention: per block = (128 q rows) x (one b,h) ----------------
__global__ __launch_bounds__(256) void attn_kernel(const u16* __restrict__ Q,
                                                   const u16* __restrict__ K,
                                                   const u16* __restrict__ V,
                                                   u16* __restrict__ O) {
  __shared__ __align__(16) u16 sK[64 * 128];        // [key][hd]
  __shared__ __align__(16) u16 sVt[128 * 64];       // [hd][key]
  __shared__ __align__(16) u16 sP[8][16 * 64];      // per-wave P tile
  const int tid = threadIdx.x;
  const int wave = tid >> 5, lane = tid & 31;
  const int laneHalf = lane & 15, hi = lane >> 4;
  const int bh = blockIdx.y;
  const size_t rowBase = (size_t)(bh >> 4) * 2048;  // batch offset in row space
  const int colBase = (bh & 15) * HEADD;            // head offset in feature space
  const int q0 = blockIdx.x * 128 + wave * 16;      // this wave's 16 q rows

  // Q fragments pinned in registers for the whole kernel (A layout)
  ABFrag qf[4];
  {
    const u16* qp = Q + (rowBase + q0 + laneHalf) * DMODEL + colBase;
#pragma unroll
    for (int kf = 0; kf < 4; kf++) {
      const u16* p = qp + kf * 32 + hi * 8;
      qf[kf].q[0] = *(const uint4*)p;
      qf[kf].q[1] = *(const uint4*)(p + 16);
    }
  }

  CFrag oa[8];
#pragma unroll
  for (int t = 0; t < 8; t++)
#pragma unroll
    for (int j = 0; j < 8; j++) oa[t].f[j] = 0.f;
  float mrow[8], lrow[8];
#pragma unroll
  for (int j = 0; j < 8; j++) { mrow[j] = -1e30f; lrow[j] = 0.f; }

  for (int kc = 0; kc < 32; kc++) {   // 32 chunks of 64 keys = S=2048
    __syncthreads();                  // previous chunk fully consumed
#if HAVE_TDM
    if (tid < 32)                     // K chunk [64][128] via TDM
      tdm_load_2d(&sK[0], K + (rowBase + kc * 64) * DMODEL + colBase, 128, 64,
                  (unsigned long long)DMODEL);
#else
    {
      int r = tid >> 2, c = (tid & 3) * 32;
      const u16* ksrc = K + (rowBase + kc * 64 + r) * DMODEL + colBase + c;
      *(uint4*)&sK[r * 128 + c]      = *(const uint4*)(ksrc);
      *(uint4*)&sK[r * 128 + c + 8]  = *(const uint4*)(ksrc + 8);
      *(uint4*)&sK[r * 128 + c + 16] = *(const uint4*)(ksrc + 16);
      *(uint4*)&sK[r * 128 + c + 24] = *(const uint4*)(ksrc + 24);
    }
#endif
    {   // V chunk transposed [128][64] (manual: TDM has no transpose)
      int r = tid >> 2, c = (tid & 3) * 32;
      const u16* vsrc = V + (rowBase + kc * 64 + r) * DMODEL + colBase + c;
#pragma unroll
      for (int g4 = 0; g4 < 4; g4++) {
        union { uint4 v; u16 h[8]; } t;
        t.v = *(const uint4*)(vsrc + g4 * 8);
#pragma unroll
        for (int j = 0; j < 8; j++) sVt[(c + g4 * 8 + j) * 64 + r] = t.h[j];
      }
    }
#if HAVE_TDM
    if (tid < 32) __builtin_amdgcn_s_wait_tensorcnt(0);
#endif
    __syncthreads();

    // scores: 16 q rows x 64 keys = 4 n-tiles x 4 k-steps of WMMA
    CFrag sc[4];
#pragma unroll
    for (int nf = 0; nf < 4; nf++) {
#pragma unroll
      for (int j = 0; j < 8; j++) sc[nf].f[j] = 0.f;
#pragma unroll
      for (int kf = 0; kf < 4; kf++) {
        ABFrag bfr;
        const u16* p = &sK[(nf * 16 + laneHalf) * 128 + kf * 32 + hi * 8];
        bfr.q[0] = *(const uint4*)p;
        bfr.q[1] = *(const uint4*)(p + 16);
        sc[nf].v = __builtin_amdgcn_wmma_f32_16x16x32_bf16(
            false, qf[kf].v, false, bfr.v, (short)0, sc[nf].v, false, false);
      }
    }

    // online softmax (row reduction across 16-lane groups)
    float corr[8];
#pragma unroll
    for (int j = 0; j < 8; j++) {
      float mx = -1e30f;
#pragma unroll
      for (int nf = 0; nf < 4; nf++) {
        sc[nf].f[j] *= ATTN_SCALE;
        mx = fmaxf(mx, sc[nf].f[j]);
      }
#pragma unroll
      for (int m = 1; m < 16; m <<= 1) mx = fmaxf(mx, __shfl_xor(mx, m, 32));
      float mn = fmaxf(mrow[j], mx);
      corr[j] = __expf(mrow[j] - mn);
      float ls = 0.f;
#pragma unroll
      for (int nf = 0; nf < 4; nf++) {
        float p = __expf(sc[nf].f[j] - mn);
        sc[nf].f[j] = p;
        ls += p;
      }
#pragma unroll
      for (int m = 1; m < 16; m <<= 1) ls += __shfl_xor(ls, m, 32);
      lrow[j] = lrow[j] * corr[j] + ls;
      mrow[j] = mn;
    }
#pragma unroll
    for (int t = 0; t < 8; t++)
#pragma unroll
      for (int j = 0; j < 8; j++) oa[t].f[j] *= corr[j];

    // write P (bf16) to this wave's LDS tile
    u16* pw = &sP[wave][0];
#pragma unroll
    for (int nf = 0; nf < 4; nf++)
#pragma unroll
      for (int j = 0; j < 8; j++)
        pw[(j + hi * 8) * 64 + nf * 16 + laneHalf] = f2bf(sc[nf].f[j]);
    __syncthreads();

    // O += P x V : 8 hd-tiles x 2 k-steps of WMMA
#pragma unroll
    for (int nt = 0; nt < 8; nt++) {
#pragma unroll
      for (int kk = 0; kk < 2; kk++) {
        ABFrag a, bfr;
        const u16* ap = &sP[wave][laneHalf * 64 + kk * 32 + hi * 8];
        a.q[0] = *(const uint4*)ap;
        a.q[1] = *(const uint4*)(ap + 16);
        const u16* bp = &sVt[(nt * 16 + laneHalf) * 64 + kk * 32 + hi * 8];
        bfr.q[0] = *(const uint4*)bp;
        bfr.q[1] = *(const uint4*)(bp + 16);
        oa[nt].v = __builtin_amdgcn_wmma_f32_16x16x32_bf16(
            false, a.v, false, bfr.v, (short)0, oa[nt].v, false, false);
      }
    }
  }

  // normalize and store
#pragma unroll
  for (int j = 0; j < 8; j++) {
    float inv = 1.f / fmaxf(lrow[j], 1e-30f);
    size_t row = rowBase + q0 + j + hi * 8;
    u16* op = O + row * DMODEL + colBase;
#pragma unroll
    for (int nt = 0; nt < 8; nt++)
      op[nt * 16 + laneHalf] = f2bf(oa[nt].f[j] * inv);
  }
}

// ---------------- workspace layout ----------------
static constexpr size_t SZ_DD  = (size_t)DMODEL * DMODEL * 2;   // 8 MB bf16
static constexpr size_t SZ_DI  = (size_t)DMODEL * DINTER * 2;   // 32 MB bf16
static constexpr size_t SZ_ACT = (size_t)NROWS * DMODEL * 2;    // 16 MB bf16
static constexpr size_t SZ_ACTF= (size_t)NROWS * DMODEL * 4;    // 32 MB f32
static constexpr size_t SZ_INT = (size_t)NROWS * DINTER * 2;    // 64 MB bf16

static constexpr size_t O_WQ    = 0;
static constexpr size_t O_WK    = O_WQ + SZ_DD;
static constexpr size_t O_WV    = O_WK + SZ_DD;
static constexpr size_t O_WO    = O_WV + SZ_DD;
static constexpr size_t O_WG    = O_WO + SZ_DD;
static constexpr size_t O_WU    = O_WG + SZ_DI;
static constexpr size_t O_WD    = O_WU + SZ_DI;
static constexpr size_t O_XLN   = O_WD + SZ_DI;
static constexpr size_t O_Q     = O_XLN + SZ_ACT;
static constexpr size_t O_K     = O_Q + SZ_ACT;
static constexpr size_t O_V     = O_K + SZ_ACT;
static constexpr size_t O_ATTN  = O_V + SZ_ACT;
static constexpr size_t O_HID   = O_ATTN + SZ_ACT;
static constexpr size_t O_Y     = O_HID + SZ_ACTF;
static constexpr size_t O_GATE  = O_Y + SZ_ACT;
static constexpr size_t O_INTER = O_GATE + SZ_INT;

extern "C" void kernel_launch(void* const* d_in, const int* in_sizes, int n_in,
                              void* d_out, int out_size, void* d_ws, size_t ws_size,
                              hipStream_t stream) {
  (void)in_sizes; (void)n_in; (void)out_size; (void)ws_size;
  const float* hs   = (const float*)d_in[0];
  const float* wq   = (const float*)d_in[1];
  const float* wk   = (const float*)d_in[2];
  const float* wv   = (const float*)d_in[3];
  const float* wo   = (const float*)d_in[4];
  const float* wg   = (const float*)d_in[5];
  const float* wu   = (const float*)d_in[6];
  const float* wd   = (const float*)d_in[7];
  const float* ln1g = (const float*)d_in[8];
  const float* ln1b = (const float*)d_in[9];
  const float* ln2g = (const float*)d_in[10];
  const float* ln2b = (const float*)d_in[11];
  float* out = (float*)d_out;

  char* ws = (char*)d_ws;
  u16* wq_bf = (u16*)(ws + O_WQ);     // all weights stored TRANSPOSED [N][K] bf16
  u16* wk_bf = (u16*)(ws + O_WK);
  u16* wv_bf = (u16*)(ws + O_WV);
  u16* wo_bf = (u16*)(ws + O_WO);
  u16* wg_bf = (u16*)(ws + O_WG);
  u16* wu_bf = (u16*)(ws + O_WU);
  u16* wd_bf = (u16*)(ws + O_WD);
  u16* xln   = (u16*)(ws + O_XLN);
  u16* qb    = (u16*)(ws + O_Q);
  u16* kb    = (u16*)(ws + O_K);
  u16* vb    = (u16*)(ws + O_V);
  u16* attnb = (u16*)(ws + O_ATTN);
  float* hid = (float*)(ws + O_HID);
  u16* yb    = (u16*)(ws + O_Y);
  u16* gateb = (u16*)(ws + O_GATE);
  u16* interb= (u16*)(ws + O_INTER);

  // weight convert + transpose: [K][N] f32 -> [N][K] bf16
  dim3 tDD(DMODEL / 32, DMODEL / 32);
  dim3 tDI(DINTER / 32, DMODEL / 32);   // K=DMODEL rows, N=DINTER cols
  dim3 tID(DMODEL / 32, DINTER / 32);   // K=DINTER rows, N=DMODEL cols
  cvt_t_kernel<<<tDD, 256, 0, stream>>>(wq, wq_bf, DMODEL, DMODEL);
  cvt_t_kernel<<<tDD, 256, 0, stream>>>(wk, wk_bf, DMODEL, DMODEL);
  cvt_t_kernel<<<tDD, 256, 0, stream>>>(wv, wv_bf, DMODEL, DMODEL);
  cvt_t_kernel<<<tDD, 256, 0, stream>>>(wo, wo_bf, DMODEL, DMODEL);
  cvt_t_kernel<<<tDI, 256, 0, stream>>>(wg, wg_bf, DMODEL, DINTER);
  cvt_t_kernel<<<tDI, 256, 0, stream>>>(wu, wu_bf, DMODEL, DINTER);
  cvt_t_kernel<<<tID, 256, 0, stream>>>(wd, wd_bf, DINTER, DMODEL);

  // LN1 -> bf16 x
  ln_kernel<<<NROWS, 256, 0, stream>>>(hs, ln1g, ln1b, xln);

  // QKV projections
  dim3 gDD(DMODEL / 256, NROWS / 128);   // (8, 32)
  gemm_kernel<0><<<gDD, 256, 0, stream>>>(xln, wq_bf, NROWS, DMODEL, DMODEL, nullptr, nullptr, nullptr, qb);
  gemm_kernel<0><<<gDD, 256, 0, stream>>>(xln, wk_bf, NROWS, DMODEL, DMODEL, nullptr, nullptr, nullptr, kb);
  gemm_kernel<0><<<gDD, 256, 0, stream>>>(xln, wv_bf, NROWS, DMODEL, DMODEL, nullptr, nullptr, nullptr, vb);

  // flash attention: grid = (q tiles, b*h)
  attn_kernel<<<dim3(2048 / 128, 2 * NHEADS), 256, 0, stream>>>(qb, kb, vb, attnb);

  // O projection + residual (f32)
  gemm_kernel<1><<<gDD, 256, 0, stream>>>(attnb, wo_bf, NROWS, DMODEL, DMODEL, hs, nullptr, hid, nullptr);

  // LN2 -> bf16 y
  ln_kernel<<<NROWS, 256, 0, stream>>>(hid, ln2g, ln2b, yb);

  // MLP: gate (silu), up (* gate), down (+ residual -> out)
  dim3 gDI(DINTER / 256, NROWS / 128);   // (32, 32)
  gemm_kernel<2><<<gDI, 256, 0, stream>>>(yb, wg_bf, NROWS, DINTER, DMODEL, nullptr, nullptr, nullptr, gateb);
  gemm_kernel<3><<<gDI, 256, 0, stream>>>(yb, wu_bf, NROWS, DINTER, DMODEL, nullptr, gateb, nullptr, interb);
  gemm_kernel<1><<<gDD, 256, 0, stream>>>(interb, wd_bf, NROWS, DMODEL, DINTER, hid, nullptr, out, nullptr);
}
